// Siamese_GeoChebyConv_26645977104605
// MI455X (gfx1250) — matrix operand order repaired and checked by reading, not compiled, not run on hardware
//
#include <hip/hip_runtime.h>

// Problem constants (fixed by the reference)
#define NNODE 200
#define NEDGE 8000
#define NFEAT 8192
#define NHID  2048
#define NCLS  512
#define KCAT1 (3 * NFEAT)   // 24576
#define KCAT2 (3 * NHID)    // 6144
#define MBAT  400           // 2 graphs x 200 nodes = 25 tiles of 16
#define SPLITK 8

typedef __attribute__((ext_vector_type(16))) __bf16 v16bf;
typedef __attribute__((ext_vector_type(8)))  float  v8f;
typedef __attribute__((ext_vector_type(4)))  unsigned int u32x4;

union OpPack { u32x4 q[2]; v16bf v; };

__device__ __forceinline__ unsigned short f2bf(float f) {
    union { float f; unsigned u; } v; v.f = f;
    unsigned r = v.u + 0x7FFFu + ((v.u >> 16) & 1u);   // round-to-nearest-even
    return (unsigned short)(r >> 16);
}

// ---------------------------------------------------------------- utilities
__global__ void zero_f32(float* __restrict__ p, int n) {
    int i = blockIdx.x * 256 + threadIdx.x;
    if (i < n) p[i] = 0.f;
}

__global__ void deg_kernel(const int* __restrict__ ei1, const float* __restrict__ ew1,
                           const int* __restrict__ ei2, const float* __restrict__ ew2,
                           float* __restrict__ deg) {
    int t = blockIdx.x * 256 + threadIdx.x;
    if (t >= 2 * NEDGE) return;
    int g = t / NEDGE, e = t - g * NEDGE;
    const int*   ei = g ? ei2 : ei1;
    const float* ew = g ? ew2 : ew1;
    atomicAdd(&deg[g * NNODE + ei[e]], ew[e]);   // row = ei[0][e]
}

__global__ void dinv_kernel(const float* __restrict__ deg, float* __restrict__ dinv) {
    int i = blockIdx.x * 256 + threadIdx.x;
    if (i >= 2 * NNODE) return;
    float d = deg[i];
    dinv[i] = (d > 0.f) ? (1.f / sqrtf(d)) : 0.f;
}

// L[g][col][row] += -dinv[row]*w*dinv[col]  so lap(z) = L @ z
__global__ void buildL_kernel(const int* __restrict__ ei1, const float* __restrict__ ew1,
                              const int* __restrict__ ei2, const float* __restrict__ ew2,
                              const float* __restrict__ dinv, float* __restrict__ Lm) {
    int t = blockIdx.x * 256 + threadIdx.x;
    if (t >= 2 * NEDGE) return;
    int g = t / NEDGE, e = t - g * NEDGE;
    const int*   ei = g ? ei2 : ei1;
    const float* ew = g ? ew2 : ew1;
    int r = ei[e], c = ei[NEDGE + e];
    float nrm = -dinv[g * NNODE + r] * ew[e] * dinv[g * NNODE + c];
    atomicAdd(&Lm[(size_t)g * NNODE * NNODE + (size_t)c * NNODE + r], nrm);
}

// fp32 [rows,F] -> bf16 into A-cat buffer at (rowBase, kOfs), leading dim ldDst
__global__ void copy_to_bf16(const float* __restrict__ src, unsigned short* __restrict__ dst,
                             int ldDst, int rowBase, int kOfs, int total, int F) {
    int idx = blockIdx.x * 256 + threadIdx.x;
    if (idx >= total) return;
    int r = idx / F, f = idx - r * F;
    dst[(size_t)(rowBase + r) * ldDst + kOfs + f] = f2bf(src[idx]);
}

// out = scale * (Lg @ in) + subCoef * sub ; optional fp32 out, always bf16 into A-cat
__global__ void lap_fuse(const float* __restrict__ Lg, const float* __restrict__ in,
                         const float* __restrict__ sub, float scale, float subCoef,
                         float* __restrict__ outF, unsigned short* __restrict__ outA,
                         int ldA, int rowBase, int kOfs, int F) {
    int f = blockIdx.x * 256 + threadIdx.x;
    if (f >= F) return;
    int i = blockIdx.y;
    float acc = 0.f;
    for (int j = 0; j < NNODE; ++j)
        acc = fmaf(Lg[i * NNODE + j], in[(size_t)j * F + f], acc);
    float val = scale * acc;
    if (sub) val = fmaf(subCoef, sub[(size_t)i * F + f], val);
    if (outF) outF[(size_t)i * F + f] = val;
    outA[(size_t)(rowBase + i) * ldA + kOfs + f] = f2bf(val);
}

// W [K][N] fp32 row-major -> Wt [N][K] bf16.
// Global->LDS staging uses CDNA5 async loads (ASYNCcnt path), then transposed ds reads.
__global__ void transpose_f32_bf16(const float* __restrict__ W, unsigned short* __restrict__ Wt,
                                   int K, int N) {
    __shared__ float tile[32][33];
    int n0 = blockIdx.x * 32, k0 = blockIdx.y * 32;
    int tx = threadIdx.x, ty = threadIdx.y;     // block (32,8)
#pragma unroll
    for (int i = 0; i < 32; i += 8) {
        unsigned ldsOff = (unsigned)(unsigned long long)&tile[ty + i][tx];
        unsigned long long ga =
            (unsigned long long)(W + (size_t)(k0 + ty + i) * N + n0 + tx);
        asm volatile("global_load_async_to_lds_b32 %0, %1, off"
                     :: "v"(ldsOff), "v"(ga) : "memory");
    }
    asm volatile("s_wait_asynccnt 0" ::: "memory");
    __syncthreads();
#pragma unroll
    for (int i = 0; i < 32; i += 8)
        Wt[(size_t)(n0 + ty + i) * K + k0 + tx] = f2bf(tile[tx][ty + i]);
}

// ------------------------------------------------------------- WMMA GEMM
// Split-K partial GEMM: P[z][MBAT][N] = A[:, kz] @ Bt[:, kz]^T for K slice z.
// One wave computes a 16(M) x 64(N) strip; double-buffered registers so the
// next k-step's loads are in flight while the current 4 WMMAs execute.
__global__ void __launch_bounds__(128)
gemm_bf16_wmma(const unsigned short* __restrict__ A,
               const unsigned short* __restrict__ Bt,
               float* __restrict__ P,
               int N, int K, int kChunk, int mTiles) {
    const int wave  = threadIdx.x >> 5;
    const int lane  = threadIdx.x & 31;
    const int mTile = blockIdx.y * 4 + wave;
    if (mTile >= mTiles) return;                 // uniform per-wave exit
    const int m0  = mTile * 16;
    const int n0  = blockIdx.x * 64;
    const int hi  = lane >> 4;
    const int l15 = lane & 15;
    const int k0   = blockIdx.z * kChunk;
    const int kEnd = k0 + kChunk;                // kChunk: multiple of 64, >= 128

    // A (16x32 bf16): lane<16 -> M=l15, K {k..k+15}; lane>=16 shifts K by 16
    const unsigned short* aRow  = A  + (size_t)(m0 + l15) * K + 16 * hi;
    // B (32x16 bf16) from Bt[N][K]: lane holds column N=n+l15, 16 contiguous K (hi -> +16)
    const unsigned short* bBase = Bt + (size_t)(n0 + l15) * K + 16 * hi;
    const size_t bS = (size_t)16 * K;

    v8f acc[4] = {};
    OpPack a0, b00, b01, b02, b03;
    OpPack a1, b10, b11, b12, b13;

    auto LDA = [&](OpPack& a, int k) {
        const u32x4* p = (const u32x4*)(aRow + k);
        a.q[0] = p[0]; a.q[1] = p[1];            // bytes +0, +16
    };
    auto LDB = [&](OpPack& b, int t, int k) {
        const u32x4* p = (const u32x4*)(bBase + bS * t + k);
        b.q[0] = p[0]; b.q[1] = p[1];            // bytes +0, +16
    };
    auto W4 = [&](OpPack& a, OpPack& b0, OpPack& b1, OpPack& b2, OpPack& b3) {
        acc[0] = __builtin_amdgcn_wmma_f32_16x16x32_bf16(false, a.v, false, b0.v, (short)0, acc[0], false, false);
        acc[1] = __builtin_amdgcn_wmma_f32_16x16x32_bf16(false, a.v, false, b1.v, (short)0, acc[1], false, false);
        acc[2] = __builtin_amdgcn_wmma_f32_16x16x32_bf16(false, a.v, false, b2.v, (short)0, acc[2], false, false);
        acc[3] = __builtin_amdgcn_wmma_f32_16x16x32_bf16(false, a.v, false, b3.v, (short)0, acc[3], false, false);
    };

    LDA(a0, k0); LDB(b00, 0, k0); LDB(b01, 1, k0); LDB(b02, 2, k0); LDB(b03, 3, k0);
    int k = k0;
    for (; k + 64 < kEnd; k += 64) {
        LDA(a1, k + 32); LDB(b10, 0, k + 32); LDB(b11, 1, k + 32); LDB(b12, 2, k + 32); LDB(b13, 3, k + 32);
        W4(a0, b00, b01, b02, b03);
        LDA(a0, k + 64); LDB(b00, 0, k + 64); LDB(b01, 1, k + 64); LDB(b02, 2, k + 64); LDB(b03, 3, k + 64);
        W4(a1, b10, b11, b12, b13);
    }
    // k == kEnd - 64 here
    LDA(a1, k + 32); LDB(b10, 0, k + 32); LDB(b11, 1, k + 32); LDB(b12, 2, k + 32); LDB(b13, 3, k + 32);
    W4(a0, b00, b01, b02, b03);
    W4(a1, b10, b11, b12, b13);

    float* Pz = P + (size_t)blockIdx.z * MBAT * N;
#pragma unroll
    for (int t = 0; t < 4; ++t) {
        const int n = n0 + t * 16 + l15;
#pragma unroll
        for (int r = 0; r < 8; ++r) {
            const int m = m0 + r + 8 * hi;       // C layout: VGPR r -> M=r (+8 for hi lanes)
            Pz[(size_t)m * N + n] = acc[t][r];
        }
    }
}

// C = sum_z P[z] + bias (ReLU opt), fp32 out + optional bf16 into next A-cat
__global__ void reduce_splitk(const float* __restrict__ P, const float* __restrict__ bias,
                              float* __restrict__ C, unsigned short* __restrict__ Cbf,
                              int ldCbf, int kOfs, int N, int total, int doRelu) {
    int idx = blockIdx.x * 256 + threadIdx.x;
    if (idx >= total) return;
    int m = idx / N, n = idx - m * N;
    float acc = bias[n];
#pragma unroll
    for (int z = 0; z < SPLITK; ++z)
        acc += P[(size_t)z * total + idx];
    if (doRelu) acc = fmaxf(acc, 0.f);
    C[idx] = acc;
    if (Cbf) Cbf[(size_t)m * ldCbf + kOfs + n] = f2bf(acc);
}

// ------------------------------------------------------------- classifier
__global__ void cls1(const float* __restrict__ O, const float* __restrict__ Wc1,
                     const float* __restrict__ bc1, float* __restrict__ h1) {
    int idx = blockIdx.x * 256 + threadIdx.x;
    if (idx >= 2 * 512 * 100) return;
    int g = idx / (512 * 100);
    int rem = idx - g * 512 * 100;
    int c = rem >> 9, r = rem & 511;
    const float* Og = O + (size_t)g * NNODE * NCLS;
    float acc = bc1[c];
    for (int j = 0; j < NNODE; ++j)
        acc = fmaf(Og[(size_t)j * NCLS + r], Wc1[c * NNODE + j], acc);
    h1[(size_t)g * 100 * 512 + (size_t)c * 512 + r] = fmaxf(acc, 0.f);
}

__global__ void cls2(const float* __restrict__ h1, const float* __restrict__ Wc2,
                     const float* __restrict__ bc2, float* __restrict__ h2) {
    int idx = blockIdx.x * 256 + threadIdx.x;
    if (idx >= 2 * 512 * 50) return;
    int g = idx / (512 * 50);
    int rem = idx - g * 512 * 50;
    int d = rem >> 9, r = rem & 511;
    const float* h1g = h1 + (size_t)g * 100 * 512;
    float acc = bc2[d];
    for (int c = 0; c < 100; ++c)
        acc = fmaf(h1g[(size_t)c * 512 + r], Wc2[d * 100 + c], acc);
    h2[(size_t)g * 50 * 512 + (size_t)d * 512 + r] = fmaxf(acc, 0.f);
}

__global__ void cls3(const float* __restrict__ h2, const float* __restrict__ Wc3,
                     const float* __restrict__ bc3, float* __restrict__ out) {
    int idx = blockIdx.x * 256 + threadIdx.x;
    if (idx >= 2 * 512) return;
    int g = idx >> 9, r = idx & 511;
    const float* h2g = h2 + (size_t)g * 50 * 512;
    float acc = bc3[0];
    for (int d = 0; d < 50; ++d)
        acc = fmaf(h2g[(size_t)d * 512 + r], Wc3[d], acc);
    out[idx] = acc;   // out1 [512] then out2 [512]
}

// ---------------------------------------------------------------- launcher
extern "C" void kernel_launch(void* const* d_in, const int* in_sizes, int n_in,
                              void* d_out, int out_size, void* d_ws, size_t ws_size,
                              hipStream_t stream) {
    (void)in_sizes; (void)n_in; (void)out_size; (void)ws_size;
    const float* x1  = (const float*)d_in[0];
    const int*   ei1 = (const int*)  d_in[1];
    const float* ew1 = (const float*)d_in[2];
    const float* x2  = (const float*)d_in[3];
    const int*   ei2 = (const int*)  d_in[4];
    const float* ew2 = (const float*)d_in[5];
    const float* W1  = (const float*)d_in[6];
    const float* b1  = (const float*)d_in[7];
    const float* W2  = (const float*)d_in[8];
    const float* b2  = (const float*)d_in[9];
    const float* Wc1 = (const float*)d_in[10];
    const float* bc1 = (const float*)d_in[11];
    const float* Wc2 = (const float*)d_in[12];
    const float* bc2 = (const float*)d_in[13];
    const float* Wc3 = (const float*)d_in[14];
    const float* bc3 = (const float*)d_in[15];
    float* out = (float*)d_out;

    char* ws = (char*)d_ws;
    size_t off = 0;
    auto alloc = [&](size_t bytes) -> char* {
        char* p = ws + off;
        off = (off + bytes + 255) & ~(size_t)255;
        return p;
    };
    unsigned short* Wt1 = (unsigned short*)alloc((size_t)NHID * KCAT1 * 2); // 100.7 MB
    unsigned short* Wt2 = (unsigned short*)alloc((size_t)NCLS * KCAT2 * 2); //   6.3 MB
    unsigned short* A1  = (unsigned short*)alloc((size_t)MBAT * KCAT1 * 2); //  19.7 MB
    unsigned short* A2  = (unsigned short*)alloc((size_t)MBAT * KCAT2 * 2); //   4.9 MB
    float* P1    = (float*)alloc((size_t)SPLITK * MBAT * NHID * 4);         //  26.2 MB
    float* P2    = (float*)alloc((size_t)SPLITK * MBAT * NCLS * 4);         //   6.6 MB
    float* T1buf = (float*)alloc((size_t)NNODE * NFEAT * 4);                //   6.6 MB
    float* H     = (float*)alloc((size_t)MBAT  * NHID  * 4);                //   3.3 MB
    float* Th1   = (float*)alloc((size_t)NNODE * NHID  * 4);                //   1.6 MB
    float* Obuf  = (float*)alloc((size_t)MBAT  * NCLS  * 4);                //   0.8 MB
    float* Lm    = (float*)alloc((size_t)2 * NNODE * NNODE * 4);
    float* deg   = (float*)alloc((size_t)2 * NNODE * 4);
    float* dinv  = (float*)alloc((size_t)2 * NNODE * 4);
    float* h1    = (float*)alloc((size_t)2 * 100 * 512 * 4);
    float* h2    = (float*)alloc((size_t)2 * 50  * 512 * 4);

    // ---- graph structure -> dense normalized Laplacians
    zero_f32<<<(2 * NNODE + 255) / 256, 256, 0, stream>>>(deg, 2 * NNODE);
    zero_f32<<<(2 * NNODE * NNODE + 255) / 256, 256, 0, stream>>>(Lm, 2 * NNODE * NNODE);
    deg_kernel<<<(2 * NEDGE + 255) / 256, 256, 0, stream>>>(ei1, ew1, ei2, ew2, deg);
    dinv_kernel<<<(2 * NNODE + 255) / 256, 256, 0, stream>>>(deg, dinv);
    buildL_kernel<<<(2 * NEDGE + 255) / 256, 256, 0, stream>>>(ei1, ew1, ei2, ew2, dinv, Lm);

    // ---- weights: fp32 [K][N] -> bf16 [N][K] (async global->LDS staging)
    transpose_f32_bf16<<<dim3(NHID / 32, KCAT1 / 32), dim3(32, 8), 0, stream>>>(W1, Wt1, KCAT1, NHID);
    transpose_f32_bf16<<<dim3(NCLS / 32, KCAT2 / 32), dim3(32, 8), 0, stream>>>(W2, Wt2, KCAT2, NCLS);

    // ---- layer 1 Chebyshev operands: A1 = [T0 | T1 | T2] per graph, bf16
    const int cp = (NNODE * NFEAT + 255) / 256;
    copy_to_bf16<<<cp, 256, 0, stream>>>(x1, A1, KCAT1, 0,     0, NNODE * NFEAT, NFEAT);
    copy_to_bf16<<<cp, 256, 0, stream>>>(x2, A1, KCAT1, NNODE, 0, NNODE * NFEAT, NFEAT);
    dim3 lg1(NFEAT / 256, NNODE);
    lap_fuse<<<lg1, 256, 0, stream>>>(Lm, x1, nullptr, 1.f, 0.f, T1buf, A1, KCAT1, 0, NFEAT, NFEAT);
    lap_fuse<<<lg1, 256, 0, stream>>>(Lm, T1buf, x1, 2.f, -1.f, nullptr, A1, KCAT1, 0, 2 * NFEAT, NFEAT);
    const float* Lm2 = Lm + NNODE * NNODE;
    lap_fuse<<<lg1, 256, 0, stream>>>(Lm2, x2, nullptr, 1.f, 0.f, T1buf, A1, KCAT1, NNODE, NFEAT, NFEAT);
    lap_fuse<<<lg1, 256, 0, stream>>>(Lm2, T1buf, x2, 2.f, -1.f, nullptr, A1, KCAT1, NNODE, 2 * NFEAT, NFEAT);

    // ---- GEMM1: [400,24576] @ [24576,2048], split-K x8, then bias+ReLU reduce
    gemm_bf16_wmma<<<dim3(NHID / 64, 7, SPLITK), 128, 0, stream>>>(
        A1, Wt1, P1, NHID, KCAT1, KCAT1 / SPLITK, MBAT / 16);
    reduce_splitk<<<(MBAT * NHID + 255) / 256, 256, 0, stream>>>(
        P1, b1, H, A2, KCAT2, 0, NHID, MBAT * NHID, 1);

    // ---- layer 2 Chebyshev operands on H
    dim3 lg2(NHID / 256, NNODE);
    lap_fuse<<<lg2, 256, 0, stream>>>(Lm, H, nullptr, 1.f, 0.f, Th1, A2, KCAT2, 0, NHID, NHID);
    lap_fuse<<<lg2, 256, 0, stream>>>(Lm, Th1, H, 2.f, -1.f, nullptr, A2, KCAT2, 0, 2 * NHID, NHID);
    const float* Hg2 = H + (size_t)NNODE * NHID;
    lap_fuse<<<lg2, 256, 0, stream>>>(Lm2, Hg2, nullptr, 1.f, 0.f, Th1, A2, KCAT2, NNODE, NHID, NHID);
    lap_fuse<<<lg2, 256, 0, stream>>>(Lm2, Th1, Hg2, 2.f, -1.f, nullptr, A2, KCAT2, NNODE, 2 * NHID, NHID);

    // ---- GEMM2: [400,6144] @ [6144,512], split-K x8, then bias reduce
    gemm_bf16_wmma<<<dim3(NCLS / 64, 7, SPLITK), 128, 0, stream>>>(
        A2, Wt2, P2, NCLS, KCAT2, KCAT2 / SPLITK, MBAT / 16);
    reduce_splitk<<<(MBAT * NCLS + 255) / 256, 256, 0, stream>>>(
        P2, b2, Obuf, nullptr, 0, 0, NCLS, MBAT * NCLS, 0);

    // ---- classifier heads on O^T
    cls1<<<(2 * 512 * 100 + 255) / 256, 256, 0, stream>>>(Obuf, Wc1, bc1, h1);
    cls2<<<(2 * 512 * 50 + 255) / 256, 256, 0, stream>>>(h1, Wc2, bc2, h2);
    cls3<<<(2 * 512 + 255) / 256, 256, 0, stream>>>(h2, Wc3, bc3, out);
}
